// JointAttention_79671643341176
// MI455X (gfx1250) — matrix operand order repaired and checked by reading, not compiled, not run on hardware
//
#include <hip/hip_runtime.h>

typedef __attribute__((ext_vector_type(8)))  _Float16 v8h;
typedef __attribute__((ext_vector_type(16))) _Float16 v16h;
typedef __attribute__((ext_vector_type(8)))  float    v8f;
typedef __attribute__((ext_vector_type(4)))  float    v4f;

#define DIM    1536
#define NH     24
#define HD     64
#define B_     2
#define NX     2048
#define NC     256
#define NTOT   2304      // NX + NC
#define QKVDIM 4608      // 3*DIM
#define SCALEQ 0.125f    // HD^-0.5 = 1/8

// ---------------------------------------------------------------------------
// WMMA helper: D = A(16x32 f16) * B(32x16 f16) + C(16x16 f32)
// ---------------------------------------------------------------------------
__device__ static inline v8f wmma_f16(v16h a, v16h b, v8f c) {
  return __builtin_amdgcn_wmma_f32_16x16x32_f16(
      /*neg_a=*/false, a, /*neg_b=*/false, b,
      /*c_mod=*/(short)0, c, /*reuse_a=*/false, /*reuse_b=*/false);
}

// A-fragment loader (wave32 16x32 f16 A layout):
// lane l (row = l&15, kb8 = (l>>4)*8):
//   halves[0..15] = A[row][2*kb8 .. 2*kb8+15] (contiguous)
// caller passes base = &A[row][kb8]; chunks are base+kb8 and base+kb8+8.
__device__ static inline v16h load_a_frag(const _Float16* base) {
  const int kb8 = ((threadIdx.x >> 4) & 1) * 8;
  v8h lo = *(const v8h*)(base + kb8);
  v8h hi = *(const v8h*)(base + kb8 + 8);
  return __builtin_shufflevector(lo, hi, 0, 1, 2, 3, 4, 5, 6, 7,
                                         8, 9, 10, 11, 12, 13, 14, 15);
}

// ---------------------------------------------------------------------------
// Async global -> LDS copy (CDNA5, tracked by ASYNCcnt). 16 bytes per lane.
// VDST VGPR carries the LDS byte address, VADDR the 64-bit global address.
// ---------------------------------------------------------------------------
__device__ static inline void async_copy_b128(unsigned lds_addr,
                                              const void* gaddr) {
  asm volatile("global_load_async_to_lds_b128 %0, %1, off"
               :: "v"(lds_addr), "v"(gaddr)
               : "memory");
}
__device__ static inline void wait_async_le4() {
  asm volatile("s_wait_asynccnt 0x4" ::: "memory");
}
__device__ static inline void wait_async_0() {
  asm volatile("s_wait_asynccnt 0x0" ::: "memory");
}

// ---------------------------------------------------------------------------
// f32 -> f16 conversion (vectorized by 8; all arrays are multiples of 8)
// ---------------------------------------------------------------------------
__global__ void cvt_f32_to_f16(const float* __restrict__ in,
                               _Float16* __restrict__ out, int n8) {
  int i = blockIdx.x * blockDim.x + threadIdx.x;
  if (i < n8) {
    v4f a = *(const v4f*)(in + (size_t)i * 8);
    v4f b = *(const v4f*)(in + (size_t)i * 8 + 4);
    v8h o;
#pragma unroll
    for (int j = 0; j < 4; ++j) {
      o[j]     = (_Float16)a[j];
      o[4 + j] = (_Float16)b[j];
    }
    *(v8h*)(out + (size_t)i * 8) = o;
  }
}

// ---------------------------------------------------------------------------
// QKV GEMM: qkv = act @ W^T + b, scattered into q [bh,N,64] (pre-scaled),
// k [bh,N,64], vT [bh,64,N].
// 8 waves/block, each wave: 32(M) x 64(N) tile (B-fragments reused 2x).
// ---------------------------------------------------------------------------
__global__ __launch_bounds__(256) void qkv_gemm(
    const _Float16* __restrict__ act,   // [B_*nseq, DIM]
    const _Float16* __restrict__ W,     // [QKVDIM, DIM] row-major
    const float*    __restrict__ bias,  // [QKVDIM]
    int nshift, int tok_off,            // nseq = 1<<nshift
    _Float16* __restrict__ qb,
    _Float16* __restrict__ kb,
    _Float16* __restrict__ vT) {
  const int lane  = threadIdx.x & 31;
  const int wave  = threadIdx.x >> 5;
  const int lcol  = lane & 15;
  const int lhalf = lane >> 4;
  const int mtile = blockIdx.x * 32;
  const int otile = blockIdx.y * 512 + wave * 64;
  const int nmask = (1 << nshift) - 1;

  const _Float16* aptr0 = act + (size_t)(mtile + lcol) * DIM + lhalf * 8;
  const _Float16* aptr1 = aptr0 + (size_t)16 * DIM;

  v8f acc[2][4] = {};

  for (int kk = 0; kk < DIM; kk += 32) {
    v16h a0 = load_a_frag(aptr0 + kk);
    v16h a1 = load_a_frag(aptr1 + kk);
#pragma unroll
    for (int t = 0; t < 4; ++t) {
      // B[k][n] = W[otile+t*16+col][kk + (l>>4)*16 + j]  (contiguous 32B)
      const _Float16* bptr =
          W + (size_t)(otile + t * 16 + lcol) * DIM + kk + lhalf * 16;
      v16h bf = *(const v16h*)bptr;
      acc[0][t] = wmma_f16(a0, bf, acc[0][t]);
      acc[1][t] = wmma_f16(a1, bf, acc[1][t]);
    }
  }

#pragma unroll
  for (int t = 0; t < 4; ++t) {
    int o     = otile + t * 16 + lcol;
    int which = o / DIM;            // 0=q 1=k 2=v (uniform within tile)
    int rem   = o - which * DIM;
    int h     = rem >> 6;
    int d     = rem & 63;
    float bo  = bias[o];
#pragma unroll
    for (int u = 0; u < 2; ++u) {
#pragma unroll
      for (int r = 0; r < 8; ++r) {
        int m   = mtile + u * 16 + r + 8 * lhalf;
        int b   = m >> nshift;
        int tok = tok_off + (m & nmask);
        int bh  = b * NH + h;
        float v = acc[u][t][r] + bo;
        if (which == 0) {
          qb[((size_t)bh * NTOT + tok) * 64 + d] = (_Float16)(v * SCALEQ);
        } else if (which == 1) {
          kb[((size_t)bh * NTOT + tok) * 64 + d] = (_Float16)v;
        } else {
          vT[((size_t)bh * 64 + d) * NTOT + tok] = (_Float16)v;
        }
      }
    }
  }
}

// ---------------------------------------------------------------------------
// Flash attention. Block = 4 waves, all on the same (b,h); each wave owns a
// 32-query stripe (two 16x16 q-tiles sharing all K/V fragments).
// K/V 32-key tiles are staged global->LDS with double-buffered async copies
// (ASYNCcnt), shared by all 4 waves: 4x less global traffic, loads overlap
// the WMMAs of the previous tile. Online softmax, P via LDS round-trip.
// out: [B_, NTOT, DIM] f16 (head-concatenated, ready for projection GEMM).
// ---------------------------------------------------------------------------
__global__ __launch_bounds__(128) void attn_kernel(
    const _Float16* __restrict__ q,    // [B_*NH, NTOT, 64], pre-scaled
    const _Float16* __restrict__ k,    // [B_*NH, NTOT, 64]
    const _Float16* __restrict__ vT,   // [B_*NH, 64, NTOT]
    _Float16* __restrict__ out) {
  __shared__ __align__(128) _Float16 kbuf[2][32 * 64];  // [key][d]
  __shared__ __align__(128) _Float16 vbuf[2][64 * 32];  // [d][key]
  __shared__ __align__(32)  _Float16 plds[4][2][16 * 32];

  const int tid   = threadIdx.x;
  const int lane  = tid & 31;
  const int wave  = tid >> 5;
  const int lcol  = lane & 15;
  const int lhalf = lane >> 4;
  const int kb16  = lhalf * 16;
  const int bh    = blockIdx.x;
  const int b     = bh / NH;
  const int h     = bh - b * NH;
  const int q0    = (blockIdx.y * 4 + wave) * 32;

  // Q fragments: a0* covers d=0..31, a1* covers d=32..63 (two q-tiles)
  const _Float16* qrow0 = q + ((size_t)bh * NTOT + q0 + lcol) * 64 + lhalf * 8;
  const _Float16* qrow1 = qrow0 + 16 * 64;
  v16h a0A = load_a_frag(qrow0);
  v16h a1A = load_a_frag(qrow0 + 32);
  v16h a0B = load_a_frag(qrow1);
  v16h a1B = load_a_frag(qrow1 + 32);

  v8f oA[4] = {}, oB[4] = {};
  float rmaxA[8], rsumA[8], rmaxB[8], rsumB[8];
#pragma unroll
  for (int r = 0; r < 8; ++r) {
    rmaxA[r] = -3.0e38f; rsumA[r] = 0.0f;
    rmaxB[r] = -3.0e38f; rsumB[r] = 0.0f;
  }

  _Float16* plA = &plds[wave][0][0];
  _Float16* plB = &plds[wave][1][0];

  const _Float16* kglob = k  + (size_t)bh * NTOT * 64;  // + j0*64 : contiguous 4KB tile
  const _Float16* vglob = vT + (size_t)bh * 64 * NTOT;  // row d: + d*NTOT + j0

  // Issue one 32-key tile (K: 4KB contiguous; V: 64 rows x 64B) as
  // 4 x b128 async copies per thread (2 for K, 2 for V).
  auto issue_tile = [&](int j0, int bufi) {
    unsigned klds = (unsigned)(size_t)&kbuf[bufi][0];
    unsigned vlds = (unsigned)(size_t)&vbuf[bufi][0];
    const char* kg = (const char*)(kglob + (size_t)j0 * 64);
    int i0 = tid, i1 = tid + 128;              // 256 chunks of 16B each
    async_copy_b128(klds + (unsigned)i0 * 16u, kg + (size_t)i0 * 16);
    async_copy_b128(klds + (unsigned)i1 * 16u, kg + (size_t)i1 * 16);
    const _Float16* vg0 =
        vglob + (size_t)(i0 >> 2) * NTOT + j0 + (i0 & 3) * 8;
    const _Float16* vg1 =
        vglob + (size_t)(i1 >> 2) * NTOT + j0 + (i1 & 3) * 8;
    async_copy_b128(vlds + (unsigned)i0 * 16u, vg0);
    async_copy_b128(vlds + (unsigned)i1 * 16u, vg1);
  };

  constexpr int NIT = NTOT / 32;   // 72
  issue_tile(0, 0);

  for (int it = 0; it < NIT; ++it) {
    if (it + 1 < NIT) {
      issue_tile((it + 1) * 32, (it + 1) & 1);
      wait_async_le4();   // loads retire in order: this tile's 4 are done
    } else {
      wait_async_0();
    }
    __syncthreads();       // all waves' copies for this tile visible

    const _Float16* kl = &kbuf[it & 1][0];
    const _Float16* vl = &vbuf[it & 1][0];

    // ---- S = Q * K^T for 32 keys (K fragments from LDS) ----
    v16h b0lo = *(const v16h*)(kl + lcol * 64 + kb16);
    v16h b0hi = *(const v16h*)(kl + lcol * 64 + 32 + kb16);
    v16h b1lo = *(const v16h*)(kl + (16 + lcol) * 64 + kb16);
    v16h b1hi = *(const v16h*)(kl + (16 + lcol) * 64 + 32 + kb16);

    v8f s0A = {}, s1A = {}, s0B = {}, s1B = {};
    s0A = wmma_f16(a0A, b0lo, s0A);
    s0A = wmma_f16(a1A, b0hi, s0A);
    s1A = wmma_f16(a0A, b1lo, s1A);
    s1A = wmma_f16(a1A, b1hi, s1A);
    s0B = wmma_f16(a0B, b0lo, s0B);
    s0B = wmma_f16(a1B, b0hi, s0B);
    s1B = wmma_f16(a0B, b1lo, s1B);
    s1B = wmma_f16(a1B, b1hi, s1B);

    // ---- online softmax (tile A) ----
#pragma unroll
    for (int r = 0; r < 8; ++r) {
      float v0 = s0A[r], v1 = s1A[r];
      float mx = fmaxf(v0, v1);
#pragma unroll
      for (int off = 1; off < 16; off <<= 1)
        mx = fmaxf(mx, __shfl_xor(mx, off, 32));
      float mnew = fmaxf(rmaxA[r], mx);
      float c    = __expf(rmaxA[r] - mnew);
      rmaxA[r]   = mnew;
      float p0   = __expf(v0 - mnew);
      float p1   = __expf(v1 - mnew);
      float ps   = p0 + p1;
#pragma unroll
      for (int off = 1; off < 16; off <<= 1)
        ps += __shfl_xor(ps, off, 32);
      rsumA[r] = rsumA[r] * c + ps;
      oA[0][r] *= c; oA[1][r] *= c; oA[2][r] *= c; oA[3][r] *= c;
      int m = r + 8 * lhalf;
      plA[m * 32 + lcol]      = (_Float16)p0;
      plA[m * 32 + 16 + lcol] = (_Float16)p1;
    }
    // ---- online softmax (tile B) ----
#pragma unroll
    for (int r = 0; r < 8; ++r) {
      float v0 = s0B[r], v1 = s1B[r];
      float mx = fmaxf(v0, v1);
#pragma unroll
      for (int off = 1; off < 16; off <<= 1)
        mx = fmaxf(mx, __shfl_xor(mx, off, 32));
      float mnew = fmaxf(rmaxB[r], mx);
      float c    = __expf(rmaxB[r] - mnew);
      rmaxB[r]   = mnew;
      float p0   = __expf(v0 - mnew);
      float p1   = __expf(v1 - mnew);
      float ps   = p0 + p1;
#pragma unroll
      for (int off = 1; off < 16; off <<= 1)
        ps += __shfl_xor(ps, off, 32);
      rsumB[r] = rsumB[r] * c + ps;
      oB[0][r] *= c; oB[1][r] *= c; oB[2][r] *= c; oB[3][r] *= c;
      int m = r + 8 * lhalf;
      plB[m * 32 + lcol]      = (_Float16)p0;
      plB[m * 32 + 16 + lcol] = (_Float16)p1;
    }

    // ---- O += P * V (K=32 keys, four d-tiles, V fragments from LDS) ----
    v16h paA = load_a_frag(plA + lcol * 32 + lhalf * 8);
    v16h paB = load_a_frag(plB + lcol * 32 + lhalf * 8);
    v16h vb0 = *(const v16h*)(vl + lcol * 32 + kb16);
    v16h vb1 = *(const v16h*)(vl + (16 + lcol) * 32 + kb16);
    v16h vb2 = *(const v16h*)(vl + (32 + lcol) * 32 + kb16);
    v16h vb3 = *(const v16h*)(vl + (48 + lcol) * 32 + kb16);
    oA[0] = wmma_f16(paA, vb0, oA[0]);
    oA[1] = wmma_f16(paA, vb1, oA[1]);
    oA[2] = wmma_f16(paA, vb2, oA[2]);
    oA[3] = wmma_f16(paA, vb3, oA[3]);
    oB[0] = wmma_f16(paB, vb0, oB[0]);
    oB[1] = wmma_f16(paB, vb1, oB[1]);
    oB[2] = wmma_f16(paB, vb2, oB[2]);
    oB[3] = wmma_f16(paB, vb3, oB[3]);
  }

  // ---- normalize and store [B, NTOT, H*64] ----
#pragma unroll
  for (int r = 0; r < 8; ++r) {
    int mA      = q0 + r + 8 * lhalf;
    float invA  = 1.0f / rsumA[r];
    _Float16* opA = out + ((size_t)b * NTOT + mA) * DIM + h * 64 + lcol;
    opA[0]  = (_Float16)(oA[0][r] * invA);
    opA[16] = (_Float16)(oA[1][r] * invA);
    opA[32] = (_Float16)(oA[2][r] * invA);
    opA[48] = (_Float16)(oA[3][r] * invA);

    int mB      = q0 + 16 + r + 8 * lhalf;
    float invB  = 1.0f / rsumB[r];
    _Float16* opB = out + ((size_t)b * NTOT + mB) * DIM + h * 64 + lcol;
    opB[0]  = (_Float16)(oB[0][r] * invB);
    opB[16] = (_Float16)(oB[1][r] * invB);
    opB[32] = (_Float16)(oB[2][r] * invB);
    opB[48] = (_Float16)(oB[3][r] * invB);
  }
}

// ---------------------------------------------------------------------------
// Output projection: out = attn @ Wproj^T + b, f32 result straight to d_out.
// 8 waves/block, each wave: 32(M) x 64(N) tile.
// ---------------------------------------------------------------------------
__global__ __launch_bounds__(256) void proj_gemm(
    const _Float16* __restrict__ attn,  // [B_*NTOT, DIM]
    const _Float16* __restrict__ W,     // [DIM, DIM] row-major
    const float*    __restrict__ bias,  // [DIM]
    int nshift, int rowoff,             // nseq = 1<<nshift; x:0, c:NX
    float* __restrict__ out) {          // [B_*nseq, DIM]
  const int lane  = threadIdx.x & 31;
  const int wave  = threadIdx.x >> 5;
  const int lcol  = lane & 15;
  const int lhalf = lane >> 4;
  const int mtile = blockIdx.x * 32;
  const int otile = blockIdx.y * 512 + wave * 64;
  const int nmask = (1 << nshift) - 1;

  int m0    = mtile + lcol;
  int b0    = m0 >> nshift;
  int arow0 = b0 * NTOT + rowoff + (m0 & nmask);
  int m1    = mtile + 16 + lcol;
  int b1    = m1 >> nshift;
  int arow1 = b1 * NTOT + rowoff + (m1 & nmask);
  const _Float16* aptr0 = attn + (size_t)arow0 * DIM + lhalf * 8;
  const _Float16* aptr1 = attn + (size_t)arow1 * DIM + lhalf * 8;

  v8f acc[2][4] = {};

  for (int kk = 0; kk < DIM; kk += 32) {
    v16h a0 = load_a_frag(aptr0 + kk);
    v16h a1 = load_a_frag(aptr1 + kk);
#pragma unroll
    for (int t = 0; t < 4; ++t) {
      const _Float16* bptr =
          W + (size_t)(otile + t * 16 + lcol) * DIM + kk + lhalf * 16;
      v16h bf = *(const v16h*)bptr;
      acc[0][t] = wmma_f16(a0, bf, acc[0][t]);
      acc[1][t] = wmma_f16(a1, bf, acc[1][t]);
    }
  }

#pragma unroll
  for (int t = 0; t < 4; ++t) {
    int o    = otile + t * 16 + lcol;
    float bo = bias[o];
#pragma unroll
    for (int u = 0; u < 2; ++u) {
#pragma unroll
      for (int r = 0; r < 8; ++r) {
        int mm = mtile + u * 16 + r + 8 * lhalf;
        out[(size_t)mm * DIM + o] = acc[u][t][r] + bo;
      }
    }
  }
}

// ---------------------------------------------------------------------------
// Host launcher
// ---------------------------------------------------------------------------
extern "C" void kernel_launch(void* const* d_in, const int* in_sizes, int n_in,
                              void* d_out, int out_size, void* d_ws,
                              size_t ws_size, hipStream_t stream) {
  const float* x       = (const float*)d_in[0];
  const float* c       = (const float*)d_in[1];
  const float* Wqkv_x  = (const float*)d_in[2];
  const float* bqkv_x  = (const float*)d_in[3];
  const float* Wqkv_c  = (const float*)d_in[4];
  const float* bqkv_c  = (const float*)d_in[5];
  const float* Wproj_x = (const float*)d_in[6];
  const float* bproj_x = (const float*)d_in[7];
  const float* Wproj_c = (const float*)d_in[8];
  const float* bproj_c = (const float*)d_in[9];

  char*  ws  = (char*)d_ws;
  size_t off = 0;
  auto carve = [&](size_t halves) {
    _Float16* p = (_Float16*)(ws + off);
    off = (off + halves * 2 + 255) & ~(size_t)255;
    return p;
  };
  _Float16* xh  = carve((size_t)B_ * NX * DIM);
  _Float16* chh = carve((size_t)B_ * NC * DIM);
  _Float16* wqx = carve((size_t)QKVDIM * DIM);
  _Float16* wqc = carve((size_t)QKVDIM * DIM);
  _Float16* wpx = carve((size_t)DIM * DIM);
  _Float16* wpc = carve((size_t)DIM * DIM);
  _Float16* qb  = carve((size_t)B_ * NH * NTOT * HD);
  _Float16* kb  = carve((size_t)B_ * NH * NTOT * HD);
  _Float16* vT  = carve((size_t)B_ * NH * HD * NTOT);
  _Float16* ao  = carve((size_t)B_ * NTOT * DIM);

  auto cvt = [&](const float* in, _Float16* o, size_t n) {
    int n8 = (int)(n / 8);
    cvt_f32_to_f16<<<(n8 + 255) / 256, 256, 0, stream>>>(in, o, n8);
  };
  cvt(x,       xh,  (size_t)B_ * NX * DIM);
  cvt(c,       chh, (size_t)B_ * NC * DIM);
  cvt(Wqkv_x,  wqx, (size_t)QKVDIM * DIM);
  cvt(Wqkv_c,  wqc, (size_t)QKVDIM * DIM);
  cvt(Wproj_x, wpx, (size_t)DIM * DIM);
  cvt(Wproj_c, wpc, (size_t)DIM * DIM);

  // QKV projections (x: 4096 rows, c: 512 rows; 32-row M tiles, 9 N-supertiles)
  qkv_gemm<<<dim3(B_ * NX / 32, QKVDIM / 512), 256, 0, stream>>>(
      xh, wqx, bqkv_x, /*nshift=*/11, /*tok_off=*/0, qb, kb, vT);
  qkv_gemm<<<dim3(B_ * NC / 32, QKVDIM / 512), 256, 0, stream>>>(
      chh, wqc, bqkv_c, /*nshift=*/8, /*tok_off=*/NX, qb, kb, vT);

  // Flash attention: 48 (b,h) blocks x 18 groups of 4 waves x 32 queries
  attn_kernel<<<dim3(B_ * NH, NTOT / 128), 128, 0, stream>>>(qb, kb, vT, ao);

  // Output projections
  float* xout = (float*)d_out;
  float* cout = xout + (size_t)B_ * NX * DIM;
  proj_gemm<<<dim3(B_ * NX / 32, DIM / 512), 256, 0, stream>>>(
      ao, wpx, bproj_x, /*nshift=*/11, /*rowoff=*/0, xout);
  proj_gemm<<<dim3(B_ * NC / 32, DIM / 512), 256, 0, stream>>>(
      ao, wpc, bproj_c, /*nshift=*/8, /*rowoff=*/NX, cout);
}
